// GCNLayer_74887049773722
// MI455X (gfx1250) — matrix operand order repaired and checked by reading, not compiled, not run on hardware
//
#include <hip/hip_runtime.h>

// ---------------------------------------------------------------------------
// GCN layer for MI455X (gfx1250, wave32, WMMA)
//   out[i,t,o] = sum_a ( (1/deg[i]) * ( sum_j adj[i,j]*X[j,t,a] + X[i,t,a] ) ) * W[a,o] + bias[o]
// Dominant GEMM (8192x8192 @ 8192x768) runs on v_wmma_f32_16x16x32_f16 with
// f32 accumulation; normalization + self-loop + W applied in f32 epilogue.
// ---------------------------------------------------------------------------

typedef _Float16 v16h   __attribute__((ext_vector_type(16)));
typedef __fp16   fp16x2 __attribute__((ext_vector_type(2)));   // cvt_pkrtz return type
typedef float    v8f    __attribute__((ext_vector_type(8)));

#define N_NODES 8192
#define T_TIMES 24
#define FEAT    32
#define NCOLS   (T_TIMES * FEAT)   // 768
#define NPW     128                 // C columns per wave
#define NT      (NPW / 16)          // 8 WMMA n-tiles per wave

union AFrag { v16h v; fp16x2 p[8]; };
union BFrag { v16h v; float4 f4[2]; };
union CFrag { v8f  v; float f[8]; };

// ---------------------------------------------------------------------------
// Kernel 1: transpose + convert X[j,c] (f32, c-contig) -> Xt[c,j] (f16, j-contig)
// so WMMA B fragments become two contiguous b128 loads per lane.
// ---------------------------------------------------------------------------
__global__ __launch_bounds__(256) void xt_transpose_kernel(
    const float* __restrict__ X, _Float16* __restrict__ Xt) {
  __shared__ float tile[32][33];
  const int j0 = blockIdx.x * 32;
  const int c0 = blockIdx.y * 32;
  const int tx = threadIdx.x;      // 0..31
  const int ty = threadIdx.y;      // 0..7
  #pragma unroll
  for (int r = ty; r < 32; r += 8)
    tile[r][tx] = X[(size_t)(j0 + r) * NCOLS + (c0 + tx)];
  __syncthreads();
  #pragma unroll
  for (int r = ty; r < 32; r += 8)
    Xt[(size_t)(c0 + r) * N_NODES + (j0 + tx)] = (_Float16)tile[tx][r];
}

// ---------------------------------------------------------------------------
// Kernel 2: deg[i] = 1 + sum_j adj[i,j]   (one 256-thread block per row)
// ---------------------------------------------------------------------------
__global__ __launch_bounds__(256) void degree_kernel(
    const float* __restrict__ adj, float* __restrict__ deg) {
  const float4* row = (const float4*)(adj + (size_t)blockIdx.x * N_NODES);
  float s = 0.0f;
  #pragma unroll
  for (int c = threadIdx.x; c < N_NODES / 4; c += 256) {
    float4 v = row[c];
    s += (v.x + v.y) + (v.z + v.w);
  }
  __shared__ float red[256];
  red[threadIdx.x] = s;
  __syncthreads();
  for (int w = 128; w > 0; w >>= 1) {
    if (threadIdx.x < w) red[threadIdx.x] += red[threadIdx.x + w];
    __syncthreads();
  }
  if (threadIdx.x == 0) deg[blockIdx.x] = red[0] + 1.0f;
}

// ---------------------------------------------------------------------------
// Kernel 3: fused  C = adj @ X  (WMMA f16->f32)  then  (C + X)*D^-1 @ W + bias
// Block: 256 threads = 8 waves. Wave w: rows [i0+16w, +16), cols [c0, c0+128).
// ---------------------------------------------------------------------------
__global__ __launch_bounds__(256) void gcn_main_kernel(
    const float* __restrict__ adj, const _Float16* __restrict__ Xt,
    const float* __restrict__ X, const float* __restrict__ W,
    const float* __restrict__ bias, const float* __restrict__ deg,
    float* __restrict__ out) {
  __shared__ float sW[FEAT * FEAT];
  __shared__ float sB[FEAT];
  __shared__ float sC[8][16][65];   // per-wave 16x64 staging, padded (bank-safe)

  const int tid  = threadIdx.x;
  const int wave = tid >> 5;
  const int lane = tid & 31;
  const int g    = lane >> 4;       // lane group (0/1)
  const int ln   = lane & 15;       // M row (A) / N col (B) within 16
  const int c0   = blockIdx.x * NPW;            // column base (t*32+a space)
  const int i0   = blockIdx.y * 128 + wave * 16; // row base for this wave

  for (int k = tid; k < FEAT * FEAT; k += 256) sW[k] = W[k];
  if (tid < FEAT) sB[tid] = bias[tid];
  __syncthreads();

  CFrag acc[NT];
  #pragma unroll
  for (int n = 0; n < NT; n++) acc[n].v = (v8f)(0.0f);

  // Per-lane adjacency row pointer (A-matrix row M = ln for both lane groups).
  const float* arow = adj + (size_t)(i0 + ln) * N_NODES;
  // Per-lane B row pointers (Xt row = column c0+16n+ln, K-contiguous).
  const _Float16* brow[NT];
  #pragma unroll
  for (int n = 0; n < NT; n++)
    brow[n] = Xt + (size_t)(c0 + 16 * n + ln) * N_NODES;

  for (int k0 = 0; k0 < N_NODES; k0 += 32) {
    // ---- Issue ALL loads for this k-step first: 4x b128 for A (f32) and
    //      8x2 b128 for the B fragments, so load latency overlaps the WMMA chain.
    const float4* pa0 = (const float4*)(arow + k0 + 16 * g);
    const float4* pa1 = (const float4*)(arow + k0 + 16 * g + 8);
    float4 x0 = pa0[0], x1 = pa0[1];
    float4 x2 = pa1[0], x3 = pa1[1];

    BFrag b[NT];
    #pragma unroll
    for (int n = 0; n < NT; n++) {
      const float4* pb = (const float4*)(brow[n] + k0 + 16 * g);
      b[n].f4[0] = pb[0];
      b[n].f4[1] = pb[1];
    }

    if (k0 + 32 < N_NODES)
      __builtin_prefetch(arow + k0 + 32, 0, 3);   // near-cache prefetch of next A slab

    // ---- A fragment: 16x32 f16. Lane (g,ln) holds row ln,
    //      K = {k0+16g .. k0+16g+15}  (contiguous, matches B)
    AFrag a;
    a.p[0] = __builtin_amdgcn_cvt_pkrtz(x0.x, x0.y);
    a.p[1] = __builtin_amdgcn_cvt_pkrtz(x0.z, x0.w);
    a.p[2] = __builtin_amdgcn_cvt_pkrtz(x1.x, x1.y);
    a.p[3] = __builtin_amdgcn_cvt_pkrtz(x1.z, x1.w);
    a.p[4] = __builtin_amdgcn_cvt_pkrtz(x2.x, x2.y);
    a.p[5] = __builtin_amdgcn_cvt_pkrtz(x2.z, x2.w);
    a.p[6] = __builtin_amdgcn_cvt_pkrtz(x3.x, x3.y);
    a.p[7] = __builtin_amdgcn_cvt_pkrtz(x3.z, x3.w);

    // ---- 8 WMMAs; B 32x16 f16: lane group g holds K=16g..16g+15 (b128 pairs).
    #pragma unroll
    for (int n = 0; n < NT; n++)
      acc[n].v = __builtin_amdgcn_wmma_f32_16x16x32_f16(
          false, a.v, false, b[n].v, (short)0, acc[n].v, false, false);
  }

  // ---- Epilogue: self-loop + D^-1 in f32, then apply W + bias via LDS.
  // C/D layout: acc[n].f[r] is row m=r+8g, col 16n+ln.
  float invd[8];
  #pragma unroll
  for (int r = 0; r < 8; r++) invd[r] = 1.0f / deg[i0 + r + 8 * g];

  #pragma unroll 1
  for (int h = 0; h < 2; h++) {       // two 64-col halves to bound LDS use
    #pragma unroll
    for (int n = 0; n < 4; n++) {
      const int nn = 4 * h + n;
      #pragma unroll
      for (int r = 0; r < 8; r++) {
        const int m = r + 8 * g;
        const int c = 16 * n + ln;    // col within this 64-wide half
        float v = acc[nn].f[r] +
                  X[(size_t)(i0 + m) * NCOLS + (c0 + 64 * h + c)]; // + I@X
        sC[wave][m][c] = v * invd[r];
      }
    }
    __syncthreads();
    // Stage 2: out = C @ W + bias over the 32-wide feature groups.
    for (int idx = lane; idx < 16 * 64; idx += 32) {
      const int m = idx >> 6;
      const int c = idx & 63;
      const int t = c >> 5;           // which 32-wide group in this half
      const int o = c & 31;
      const float* crow = &sC[wave][m][t * 32];
      float s = sB[o];
      #pragma unroll
      for (int a2 = 0; a2 < FEAT; a2++) s += crow[a2] * sW[a2 * FEAT + o];
      out[(size_t)(i0 + m) * NCOLS + (c0 + 64 * h + c)] = s;
    }
    __syncthreads();
  }
}

// ---------------------------------------------------------------------------
extern "C" void kernel_launch(void* const* d_in, const int* in_sizes, int n_in,
                              void* d_out, int out_size, void* d_ws, size_t ws_size,
                              hipStream_t stream) {
  const float* node_feats = (const float*)d_in[0];  // [8192, 24, 32] f32
  const float* adj        = (const float*)d_in[1];  // [8192, 8192]  f32
  const float* weight     = (const float*)d_in[2];  // [32, 32]      f32
  const float* bias       = (const float*)d_in[3];  // [32]          f32
  float* out = (float*)d_out;                       // [8192, 24, 32] f32

  // Workspace: Xt (f16, 768*8192 = 12.6 MB) then deg (f32, 32 KB).
  _Float16* Xt = (_Float16*)d_ws;
  float* deg = (float*)((char*)d_ws + (size_t)NCOLS * N_NODES * sizeof(_Float16));

  xt_transpose_kernel<<<dim3(N_NODES / 32, NCOLS / 32), dim3(32, 8), 0, stream>>>(
      node_feats, Xt);
  degree_kernel<<<N_NODES, 256, 0, stream>>>(adj, deg);
  gcn_main_kernel<<<dim3(NCOLS / NPW, N_NODES / 128), 256, 0, stream>>>(
      adj, Xt, node_feats, weight, bias, deg, out);
}
